// DySample_43817256354064
// MI455X (gfx1250) — compile-verified, matrix operand order checked by reading, not run on hardware
//
#include <hip/hip_runtime.h>
#include <hip/hip_bf16.h>
#include <math.h>

typedef __bf16 bf16_t;
typedef __attribute__((ext_vector_type(16))) __bf16        v16bf;
typedef __attribute__((ext_vector_type(8)))  float         v8f;
typedef __attribute__((ext_vector_type(4)))  float         v4f;
typedef __attribute__((ext_vector_type(4)))  unsigned int  uint4v;

#define B_   16
#define C_   64
#define H_   80
#define W_   80
#define HS   160
#define WS   160
#define NPIX (HS * WS)                    // 25600 pixels per image
#define NTOT (B_ * NPIX)                  // 409600 GEMM rows
#define S_BYTES ((size_t)NTOT * C_ * 2)   // staged bf16 activations

union FragBF { v16bf v; uint4v u[2]; };

// ---------------------------------------------------------------------------
// Kernel 0: pack conv_w into fragment-ordered bf16 (ISA 16-bit B 32x16 lane
// layout) and fold BN into (scale, shift).  8 KB + 512 B, runs once.
// ---------------------------------------------------------------------------
__global__ __launch_bounds__(256) void dysample_pack_kernel(
    const float* __restrict__ conv_w,
    const float* __restrict__ gamma, const float* __restrict__ beta,
    const float* __restrict__ mean,  const float* __restrict__ var,
    bf16_t* __restrict__ Wf, float* __restrict__ bnp)
{
  const int t    = threadIdx.x;          // (ct, kb, lane) = exactly 256 combos
  const int lane = t & 31;
  const int m    = lane & 15;
  const int c0   = (lane >= 16) ? 8 : 0;
  const int ct   = t >> 6;
  const int kb   = (t >> 5) & 1;
  const int o    = ct * 16 + m;

  const float* wrow = conv_w + o * 64 + kb * 32;
  unsigned int* dst = (unsigned int*)(Wf + (size_t)t * 16);
#pragma unroll
  for (int h = 0; h < 2; ++h) {          // elems 0..7 = K c0.., 8..15 = K c0+16..
    const int kbase = c0 + h * 16;
#pragma unroll
    for (int j = 0; j < 8; j += 2) {
      bf16_t x0 = (bf16_t)wrow[kbase + j];
      bf16_t x1 = (bf16_t)wrow[kbase + j + 1];
      unsigned short u0 = __builtin_bit_cast(unsigned short, x0);
      unsigned short u1 = __builtin_bit_cast(unsigned short, x1);
      dst[(h * 8 + j) >> 1] = (unsigned int)u0 | ((unsigned int)u1 << 16);
    }
  }
  if (t < 64) {
    float sc = gamma[t] * rsqrtf(var[t] + 1e-5f);
    bnp[t]      = sc;
    bnp[64 + t] = beta[t] - mean[t] * sc;
  }
}

// ---------------------------------------------------------------------------
// 1-D tap expansion: grid-sample taps (160 grid) composed with the bilinear
// upsample (80 grid) -> 4 (index, weight) pairs; OOB validity folded in.
// ---------------------------------------------------------------------------
__device__ __forceinline__ void taps1d(float coord, int idx[4], float wt[4]) {
  float f0 = floorf(coord);
  int   i0 = (int)f0;
  float w1 = coord - f0;
  float tw[2] = {1.0f - w1, w1};
  int   ti[2] = {i0, i0 + 1};
#pragma unroll
  for (int k = 0; k < 2; ++k) {
    int   it    = ti[k];
    float valid = (it >= 0 && it < 160) ? tw[k] : 0.0f;
    float c  = fminf(fmaxf((float)it * 0.5f - 0.25f, 0.0f), 79.0f);
    float cf = floorf(c);
    int   c0 = (int)cf;
    int   c1 = c0 + 1; if (c1 > 79) c1 = 79;
    float wc = c - cf;
    idx[2 * k]     = c0; wt[2 * k]     = valid * (1.0f - wc);
    idx[2 * k + 1] = c1; wt[2 * k + 1] = valid * wc;
  }
}

// ---------------------------------------------------------------------------
// Kernel 1: offset/mask 1x1 convs + fused (upsample o grid_sample) * mask.
// Stages the block's 256x64 bf16 tile in LDS, then streams 32 KB contiguously
// to the workspace via global_store_async_from_lds_b128 (ASYNCcnt path).
// ---------------------------------------------------------------------------
__global__ __launch_bounds__(256) void dysample_sample_kernel(
    const float* __restrict__ x,
    const float* __restrict__ offset_w, const float* __restrict__ offset_b,
    const float* __restrict__ mask_w,   const float* __restrict__ mask_b,
    bf16_t* __restrict__ S)
{
  __shared__ unsigned int smem[256 * 36];          // padded rows (36 dwords)
  const int tid = threadIdx.x;
  const int n   = blockIdx.x * 256 + tid;          // grid is exact
  int b  = n / NPIX;
  int p  = n - b * NPIX;
  int hs = p / WS;
  int ws = p - hs * WS;
  int h = hs >> 1, w = ws >> 1;
  int sub = ((hs & 1) << 1) | (ws & 1);            // pixel-shuffle sub-position

  // 1x1 convs at source pixel: offset_x, offset_y, mask (3 dots over 64 ch)
  const float* xp  = x + (size_t)b * C_ * (H_ * W_) + h * W_ + w;
  const float* ow0 = offset_w + sub * C_;
  const float* ow1 = offset_w + (4 + sub) * C_;
  const float* mw  = mask_w + sub * C_;
  float ox = offset_b[sub], oy = offset_b[4 + sub];
  float mk = mask_b[sub];
#pragma unroll 8
  for (int c = 0; c < C_; ++c) {
    float xv = xp[c * (H_ * W_)];
    ox = fmaf(ow0[c], xv, ox);
    oy = fmaf(ow1[c], xv, oy);
    mk = fmaf(mw[c],  xv, mk);
  }
  mk = 1.0f / (1.0f + __expf(-mk));                // sigmoid(mask)

  float gx = -1.0f + (float)ws * (2.0f / 159.0f) + ox;
  float gy = -1.0f + (float)hs * (2.0f / 159.0f) + oy;
  float ix = ((gx + 1.0f) * (float)WS - 1.0f) * 0.5f;
  float iy = ((gy + 1.0f) * (float)HS - 1.0f) * 0.5f;

  int ci[4]; float cw[4];
  int ri[4]; float rw[4];
  taps1d(ix, ci, cw);
  taps1d(iy, ri, rw);

  int   off[16];
  float wgt[16];
#pragma unroll
  for (int i = 0; i < 4; ++i)
#pragma unroll
    for (int j = 0; j < 4; ++j) {
      off[i * 4 + j] = ri[i] * W_ + ci[j];
      wgt[i * 4 + j] = rw[i] * cw[j];
    }

  const float* plane0 = x + (size_t)b * C_ * (H_ * W_);
  unsigned int* lrow = smem + tid * 36;
#pragma unroll 2
  for (int c = 0; c < C_; c += 2) {
    const float* p0 = plane0 + (size_t)c * (H_ * W_);
    const float* p1 = p0 + (H_ * W_);
    float s0 = 0.0f, s1 = 0.0f;
#pragma unroll
    for (int t = 0; t < 16; ++t) {
      s0 = fmaf(wgt[t], p0[off[t]], s0);
      s1 = fmaf(wgt[t], p1[off[t]], s1);
    }
    s0 *= mk; s1 *= mk;
    bf16_t b0 = (bf16_t)s0, b1 = (bf16_t)s1;
    unsigned short u0 = __builtin_bit_cast(unsigned short, b0);
    unsigned short u1 = __builtin_bit_cast(unsigned short, b1);
    lrow[c >> 1] = (unsigned int)u0 | ((unsigned int)u1 << 16);
  }

  __syncthreads();

  // Contiguous 32 KB writeback, LDS -> global, async path (ASYNCcnt).
  char* Sblock = (char*)S + (size_t)blockIdx.x * 256 * C_ * 2;
#pragma unroll
  for (int k = 0; k < 8; ++k) {
    int c  = k * 256 + tid;                        // 16-B chunk id (coalesced)
    int r  = c >> 3;                               // source pixel row in LDS
    int j0 = (c & 7) * 4;                          // dword offset within row
    unsigned loff = (unsigned)(uintptr_t)&smem[r * 36 + j0];
    unsigned long long g = (unsigned long long)(uintptr_t)(Sblock + (size_t)c * 16);
    asm volatile("global_store_async_from_lds_b128 %0, %1, off"
                 :: "v"(g), "v"(loff) : "memory");
  }
  asm volatile("s_wait_asynccnt 0" ::: "memory");
}

// ---------------------------------------------------------------------------
// Kernel 2: [NTOT,64] x W^T GEMM via v_wmma_f32_16x16x32_bf16 + BN + SiLU.
// 8 waves/block, 4 row-tiles/wave (32 WMMA per wave); B fragments registered
// once; output transposed through a wave-private LDS slab for 16-B stores.
// ---------------------------------------------------------------------------
__global__ __launch_bounds__(256) void dysample_gemm_kernel(
    const bf16_t* __restrict__ S, const bf16_t* __restrict__ Wf,
    const float* __restrict__ bnp, float* __restrict__ out)
{
  __shared__ float smem[8 * 64 * 20];              // 8 wave slabs, padded rows
  const int  lane = threadIdx.x & 31;
  const int  wave = threadIdx.x >> 5;
  const bool hi   = lane >= 16;
  const int  c0   = hi ? 8 : 0;                    // ISA A/B lane K base
  const int  m    = lane & 15;
  const int  rofs = hi ? 8 : 0;
  float* slab = smem + wave * (64 * 20);

  // B fragments (pre-packed, fragment-ordered): pure b128 loads.
  FragBF bf[4][2];
#pragma unroll
  for (int ct = 0; ct < 4; ++ct)
#pragma unroll
    for (int kb = 0; kb < 2; ++kb) {
      const uint4v* p =
          (const uint4v*)(Wf + (size_t)(((ct * 2 + kb) * 32 + lane)) * 16);
      bf[ct][kb].u[0] = p[0];
      bf[ct][kb].u[1] = p[1];
    }

  float scv[4], shv[4];
#pragma unroll
  for (int ct = 0; ct < 4; ++ct) {
    int o = ct * 16 + m;
    scv[ct] = bnp[o];
    shv[ct] = bnp[64 + o];
  }

  const int    blockStart = blockIdx.x * 512;      // 512 rows, same image
  const int    bImg  = blockStart / NPIX;
  const size_t outImg = (size_t)bImg * C_ * NPIX;

#pragma unroll
  for (int t = 0; t < 4; ++t) {
    const int rowBase = blockStart + wave * 64 + t * 16;
    const int pBase   = rowBase - bImg * NPIX;

    // A fragments: lane holds K = {c0..c0+7, c0+16..c0+23} per 32-K block
    const bf16_t* arow = S + (size_t)(rowBase + m) * 64;
    FragBF a0, a1;
    a0.u[0] = *(const uint4v*)(arow + c0);
    a0.u[1] = *(const uint4v*)(arow + c0 + 16);
    a1.u[0] = *(const uint4v*)(arow + 32 + c0);
    a1.u[1] = *(const uint4v*)(arow + 32 + c0 + 16);

#pragma unroll
    for (int ct = 0; ct < 4; ++ct) {
      v8f acc = {0.f, 0.f, 0.f, 0.f, 0.f, 0.f, 0.f, 0.f};
      acc = __builtin_amdgcn_wmma_f32_16x16x32_bf16(false, a0.v, false,
              bf[ct][0].v, (short)0, acc, false, false);
      acc = __builtin_amdgcn_wmma_f32_16x16x32_bf16(false, a1.v, false,
              bf[ct][1].v, (short)0, acc, false, false);

      // BN + SiLU, stash into this wave's LDS slab (rows contiguous per ch)
      const int   o  = ct * 16 + m;
      const float sc = scv[ct], sh = shv[ct];
      v4f ya, yb;
#pragma unroll
      for (int r = 0; r < 4; ++r) {
        float y0 = fmaf(acc[r],     sc, sh);
        float y1 = fmaf(acc[r + 4], sc, sh);
        ya[r] = y0 / (1.0f + __expf(-y0));
        yb[r] = y1 / (1.0f + __expf(-y1));
      }
      float* dst = slab + o * 20 + rofs;
      *(v4f*)dst       = ya;
      *(v4f*)(dst + 4) = yb;
    }

    // Coalesced writeout: 16-B chunks, 64-B contiguous per channel segment.
#pragma unroll
    for (int k = 0; k < 8; ++k) {
      int chunk = k * 32 + lane;
      int o = chunk >> 2, q = chunk & 3;
      v4f v = *(const v4f*)(slab + o * 20 + q * 4);
      *(v4f*)(out + outImg + (size_t)o * NPIX + pBase + q * 4) = v;
    }
  }
}

extern "C" void kernel_launch(void* const* d_in, const int* in_sizes, int n_in,
                              void* d_out, int out_size, void* d_ws, size_t ws_size,
                              hipStream_t stream) {
  const float* x        = (const float*)d_in[0];
  const float* offset_w = (const float*)d_in[1];
  const float* offset_b = (const float*)d_in[2];
  const float* mask_w   = (const float*)d_in[3];
  const float* mask_b   = (const float*)d_in[4];
  const float* conv_w   = (const float*)d_in[5];
  const float* gamma    = (const float*)d_in[6];
  const float* beta     = (const float*)d_in[7];
  const float* mean     = (const float*)d_in[8];
  const float* var      = (const float*)d_in[9];

  bf16_t* S   = (bf16_t*)d_ws;                                  // ~52 MB
  bf16_t* Wf  = (bf16_t*)((char*)d_ws + S_BYTES);               // 8 KB
  float*  bnp = (float*)((char*)d_ws + S_BYTES + 8192);         // 512 B

  dysample_pack_kernel<<<1, 256, 0, stream>>>(
      conv_w, gamma, beta, mean, var, Wf, bnp);
  dysample_sample_kernel<<<NTOT / 256, 256, 0, stream>>>(
      x, offset_w, offset_b, mask_w, mask_b, S);
  dysample_gemm_kernel<<<NTOT / 512, 256, 0, stream>>>(
      S, Wf, bnp, (float*)d_out);
}
